// RPNHead_31885837205765
// MI455X (gfx1250) — compile-verified, matrix operand order checked by reading, not run on hardware
//
#include <hip/hip_runtime.h>
#include <hip/hip_bf16.h>
#include <stdint.h>

typedef __attribute__((ext_vector_type(16))) __bf16 v16bf;
typedef __attribute__((ext_vector_type(8)))  float  v8f;

#define ATOT 261888                 // total anchors per batch image (sum 3*s^2)
#define WPACK_ELEMS (9*8*32*32*16)  // taps * kc * ntile * lane * 16  = 1,179,648
#define W18_ELEMS   (16*2*32*16)    // kk * ntile2 * lane * 16        = 16,384
#define WS_W18_OFF  (WPACK_ELEMS*2)             // bytes
#define WS_B18_OFF  (WS_W18_OFF + W18_ELEMS*2)  // bytes

// ---------------- weight repack: w_share (3,3,256,512) HWIO -> bf16 B-fragments ----
__global__ void pack_wshare_kernel(const float* __restrict__ w, __bf16* __restrict__ wp) {
  int i = blockIdx.x * 256 + threadIdx.x;
  if (i >= WPACK_ELEMS) return;
  int t    = i & 15;
  int lane = (i >> 4) & 31;
  int nt   = (i >> 9) & 31;
  int kc   = (i >> 14) & 7;
  int tap  = i >> 17;
  int c = kc * 32 + (lane >> 4) * 16 + t;   // input channel (K)
  int n = nt * 16 + (lane & 15);            // output channel (N)
  wp[i] = (__bf16)w[((size_t)tap * 256 + c) * 512 + n];
}

// ---------------- head weights: (512,6) + (512,12) -> bf16 B-fragments, N padded to 32 ----
__global__ void pack_w18_kernel(const float* __restrict__ wcls, const float* __restrict__ wbox,
                                __bf16* __restrict__ wp) {
  int i = blockIdx.x * 256 + threadIdx.x;
  if (i >= W18_ELEMS) return;
  int t    = i & 15;
  int lane = (i >> 4) & 31;
  int nt2  = (i >> 9) & 1;
  int kk   = i >> 10;
  int k = kk * 32 + (lane >> 4) * 16 + t;
  int n = nt2 * 16 + (lane & 15);
  float v = 0.0f;
  if (n < 6)       v = wcls[k * 6 + n];
  else if (n < 18) v = wbox[k * 12 + (n - 6)];
  wp[i] = (__bf16)v;
}

__global__ void pack_bias18_kernel(const float* __restrict__ bcls, const float* __restrict__ bbox,
                                   float* __restrict__ b18) {
  int n = threadIdx.x;
  if (n >= 32) return;
  float v = 0.0f;
  if (n < 6)       v = bcls[n];
  else if (n < 18) v = bbox[n - 6];
  b18[n] = v;
}

// ---------------- fused conv3x3(+bias,relu) + 1x1 heads + softmax + scatter ----------
// One WG = 256 threads = 8 wave32, owns 64 pixels x 512 channels.
__global__ __launch_bounds__(256, 1)
void rpn_level_kernel(const float* __restrict__ x,
                      const __bf16* __restrict__ wpack,
                      const float* __restrict__ bshare,
                      const __bf16* __restrict__ w18,
                      const float* __restrict__ b18,
                      float* __restrict__ out,
                      int s, int lvlbase)
{
  extern __shared__ char smem[];
  const int tid   = threadIdx.x;
  const int lane  = tid & 31;
  const int wave  = tid >> 5;
  const int mrow  = lane & 15;
  const int khalf = lane >> 4;

  const int Wd  = (s >= 64) ? 64 : s;               // pixels per row of the tile
  const int wsh = (Wd == 64) ? 6 : ((Wd == 32) ? 5 : 4);
  const int R   = 64 >> wsh;                        // rows per tile
  const int PC  = Wd + 2;                           // padded patch cols
  const int PR  = R + 2;                            // padded patch rows

  const int ss  = s * s;
  const int p0  = blockIdx.x * 64;                  // first pixel of this tile
  const int img = p0 / ss;
  const int y0  = (p0 - img * ss) / s;
  const int x0  = p0 % s;

  __bf16* patch = (__bf16*)smem;                    // [PR][PC][256] bf16

  // ---- Phase 1: stage zero-padded input halo into LDS as bf16 ----
  {
    const int units = PR * PC * 32;                 // 8 channels per unit
    for (int u = tid; u < units; u += 256) {
      int ch8  = u & 31;
      int cell = u >> 5;
      int pc   = cell % PC;
      int pr   = cell / PC;
      int iy   = y0 - 1 + pr;
      int ix   = x0 - 1 + pc;
      union { __bf16 h[8]; uint4 q; } pk;
      if (iy >= 0 && iy < s && ix >= 0 && ix < s) {
        const float4* g =
            (const float4*)(x + (((size_t)img * ss + (size_t)iy * s + ix) * 256 + ch8 * 8));
        float4 f0 = g[0];
        float4 f1 = g[1];
        pk.h[0] = (__bf16)f0.x; pk.h[1] = (__bf16)f0.y;
        pk.h[2] = (__bf16)f0.z; pk.h[3] = (__bf16)f0.w;
        pk.h[4] = (__bf16)f1.x; pk.h[5] = (__bf16)f1.y;
        pk.h[6] = (__bf16)f1.z; pk.h[7] = (__bf16)f1.w;
      } else {
        pk.q = make_uint4(0u, 0u, 0u, 0u);
      }
      *(uint4*)(patch + ((size_t)cell * 256 + ch8 * 8)) = pk.q;
    }
  }
  __syncthreads();

  // ---- Phase 2: implicit-GEMM 3x3 conv, M=64 px, N=512 ch, K=9*256 ----
  // Wave w owns N-tiles [4w, 4w+4). Per-lane byte offsets into the LDS patch (A layout:
  // row M = lane%16, K half = lane/16, 16 contiguous bf16 per lane).
  int aoff[4];
#pragma unroll
  for (int mt = 0; mt < 4; ++mt) {
    int m  = mt * 16 + mrow;
    int ty = m >> wsh;
    int tx = m & (Wd - 1);
    aoff[mt] = (((ty + 1) * PC + (tx + 1)) * 256 + khalf * 16) * 2;
  }
  const __bf16* wlane = wpack + ((size_t)(wave * 4) * 512 + lane * 16);

  v8f acc[4][4];
#pragma unroll
  for (int mt = 0; mt < 4; ++mt)
#pragma unroll
    for (int nt = 0; nt < 4; ++nt)
      acc[mt][nt] = (v8f){0.f, 0.f, 0.f, 0.f, 0.f, 0.f, 0.f, 0.f};

#pragma unroll 1
  for (int tap = 0; tap < 9; ++tap) {
    const int dy = tap / 3 - 1;
    const int dx = tap % 3 - 1;
    const int adelta = (dy * PC + dx) * 512;                 // bytes
    const __bf16* wtap = wlane + (size_t)tap * 8 * 16384;    // 8 kc * 32 frag * 512 el
#pragma unroll 2
    for (int kc = 0; kc < 8; ++kc) {
      v16bf afrag[4], bfrag[4];
#pragma unroll
      for (int mt = 0; mt < 4; ++mt)
        afrag[mt] = *(const v16bf*)((const char*)patch + (aoff[mt] + adelta + kc * 64));
#pragma unroll
      for (int nt = 0; nt < 4; ++nt)
        bfrag[nt] = *(const v16bf*)(wtap + (size_t)kc * 16384 + nt * 512);
#pragma unroll
      for (int mt = 0; mt < 4; ++mt)
#pragma unroll
        for (int nt = 0; nt < 4; ++nt)
          acc[mt][nt] = __builtin_amdgcn_wmma_f32_16x16x32_bf16(
              false, afrag[mt], false, bfrag[nt], (short)0, acc[mt][nt], false, false);
    }
  }
  __syncthreads();   // all patch reads complete before LDS reuse

  // ---- Phase 3: bias + ReLU, deposit S = relu(conv) as bf16 [64][512] in LDS ----
  __bf16* S = (__bf16*)smem;
#pragma unroll
  for (int nt = 0; nt < 4; ++nt) {
    const int   nch = wave * 64 + nt * 16 + mrow;   // C/D layout: N = lane%16
    const float bz  = bshare[nch];
#pragma unroll
    for (int mt = 0; mt < 4; ++mt) {
#pragma unroll
      for (int r = 0; r < 8; ++r) {                 // C/D layout: M = r + 8*(lane/16)
        float v = acc[mt][nt][r] + bz;
        v = v > 0.f ? v : 0.f;
        S[(size_t)(mt * 16 + khalf * 8 + r) * 512 + nch] = (__bf16)v;
      }
    }
  }
  __syncthreads();

  // ---- Phase 4: 1x1 heads via WMMA, K=512, N=32 (6 cls + 12 box + 14 pad) ----
  float* out18 = (float*)(smem + 65536);            // [64][32] f32
  {
    const int mt2 = wave & 3;
    const int nt2 = wave >> 2;
    v8f a2 = (v8f){0.f, 0.f, 0.f, 0.f, 0.f, 0.f, 0.f, 0.f};
#pragma unroll
    for (int kk = 0; kk < 16; ++kk) {
      v16bf af = *(const v16bf*)(S + ((size_t)(mt2 * 16 + mrow) * 512 + kk * 32 + khalf * 16));
      v16bf bf = *(const v16bf*)(w18 + (((size_t)(kk * 2 + nt2) * 32 + lane) * 16));
      a2 = __builtin_amdgcn_wmma_f32_16x16x32_bf16(
          false, af, false, bf, (short)0, a2, false, false);
    }
    const float bz = b18[nt2 * 16 + mrow];
#pragma unroll
    for (int r = 0; r < 8; ++r)
      out18[(mt2 * 16 + khalf * 8 + r) * 32 + nt2 * 16 + mrow] = a2[r] + bz;
  }
  __syncthreads();

  // ---- Phase 5: softmax over anchor pairs + scatter into concatenated outputs ----
  if (tid < 192) {                                  // 64 px * 3 anchors
    const int px  = tid / 3;
    const int anc = tid - px * 3;
    const float* o  = out18 + px * 32;
    const float l0  = o[anc * 2 + 0];
    const float l1  = o[anc * 2 + 1];
    const float mx  = fmaxf(l0, l1);
    const float e0  = __expf(l0 - mx);
    const float e1  = __expf(l1 - mx);
    const float inv = 1.0f / (e0 + e1);

    const int  p    = p0 + px;
    const int  nimg = p / ss;
    const int  pin  = p - nimg * ss;
    const long a    = (long)lvlbase + (long)pin * 3 + anc;
    const long row  = (long)nimg * ATOT + a;

    out[row * 2 + 0] = l0;                          // class_logit
    out[row * 2 + 1] = l1;
    float* prob = out + (long)2 * ATOT * 2;         // class_prob
    prob[row * 2 + 0] = e0 * inv;
    prob[row * 2 + 1] = e1 * inv;
    float* box = out + (long)4 * ATOT * 2;          // box_pred
    const float* ob = o + 6 + anc * 4;
    box[row * 4 + 0] = ob[0];
    box[row * 4 + 1] = ob[1];
    box[row * 4 + 2] = ob[2];
    box[row * 4 + 3] = ob[3];
  }
}

// ---------------- host launcher ----------------
extern "C" void kernel_launch(void* const* d_in, const int* in_sizes, int n_in,
                              void* d_out, int out_size, void* d_ws, size_t ws_size,
                              hipStream_t stream) {
  (void)in_sizes; (void)n_in; (void)out_size; (void)ws_size;
  const float* w_share = (const float*)d_in[5];
  const float* b_share = (const float*)d_in[6];
  const float* w_cls   = (const float*)d_in[7];
  const float* b_cls   = (const float*)d_in[8];
  const float* w_box   = (const float*)d_in[9];
  const float* b_box   = (const float*)d_in[10];

  __bf16* wpack = (__bf16*)d_ws;
  __bf16* w18   = (__bf16*)((char*)d_ws + WS_W18_OFF);
  float*  b18   = (float*)((char*)d_ws + WS_B18_OFF);

  pack_wshare_kernel<<<(WPACK_ELEMS + 255) / 256, 256, 0, stream>>>(w_share, wpack);
  pack_w18_kernel<<<(W18_ELEMS + 255) / 256, 256, 0, stream>>>(w_cls, w_box, w18);
  pack_bias18_kernel<<<1, 32, 0, stream>>>(b_cls, b_box, b18);

  const int sizes[5]   = {256, 128, 64, 32, 16};
  const int lvlbase[5] = {0, 196608, 245760, 258048, 261120};
  for (int i = 0; i < 5; ++i) {
    const int s = sizes[i];
    const int blocks = 2 * s * s / 64;
    const int Wd = (s >= 64) ? 64 : s;
    const int R  = 64 / Wd;
    size_t lds = (size_t)(R + 2) * (Wd + 2) * 256 * 2;   // bf16 patch
    if (lds < 73728) lds = 73728;                         // S (64x512 bf16) + out18 (64x32 f32)
    rpn_level_kernel<<<blocks, 256, lds, stream>>>(
        (const float*)d_in[i], wpack, b_share, w18, b18, (float*)d_out, s, lvlbase[i]);
  }
}
